// SASA_69355131896242
// MI455X (gfx1250) — matrix operand order repaired and checked by reading, not compiled, not run on hardware
//
#include <hip/hip_runtime.h>
#include <hip/hip_bf16.h>

typedef __attribute__((ext_vector_type(16))) _Float16 v16h;
typedef __attribute__((ext_vector_type(8)))  float    v8f;
typedef __attribute__((ext_vector_type(4)))  unsigned int u32x4;
typedef __attribute__((ext_vector_type(8)))  int      i32x8;
typedef __attribute__((ext_vector_type(4)))  int      i32x4;

#define B_     8
#define C_     256
#define H_     128
#define W_     128
#define TOK_   4
#define HH_    32
#define WW_    32
#define NW_    1024
#define HEADS_ 8
#define HD_    32
#define PP_    16
#define NPIX_  16384

#define WMMA16(A, Bv, Cv) \
  __builtin_amdgcn_wmma_f32_16x16x32_f16(false, (A), false, (Bv), (short)0, (Cv), false, false)

// ---------------- 4x4 average pool: x (b,n,c) -> wf (b,nw,c) ----------------
__global__ void k_pool(const float* __restrict__ x, float* __restrict__ wf) {
  int n = blockIdx.x, b = blockIdx.y, c = threadIdx.x;
  int wy = n / WW_, wx = n % WW_;
  float s = 0.f;
#pragma unroll
  for (int p = 0; p < PP_; ++p) {
    int y = wy * TOK_ + (p >> 2), xx = wx * TOK_ + (p & 3);
    s += x[((size_t)(b * NPIX_ + y * W_ + xx)) * C_ + c];
  }
  wf[((size_t)(b * NW_ + n)) * C_ + c] = s * (1.f / 16.f);
}

// ------- affinity + softmax(9) + fold-scatter of pix*aff and aff mass -------
__global__ void k_aff(const float* __restrict__ x, const float* __restrict__ wf,
                      float* __restrict__ aff, float* __restrict__ st,
                      float* __restrict__ affsum) {
  __shared__ float pixs[PP_][C_ + 1];
  __shared__ float wfus[9][C_ + 1];
  __shared__ float affs[PP_][9];
  int n = blockIdx.x, b = blockIdx.y, t = threadIdx.x;
  int wy = n / WW_, wx = n % WW_;
#pragma unroll
  for (int p = 0; p < PP_; ++p) {
    int y = wy * TOK_ + (p >> 2), xx = wx * TOK_ + (p & 3);
    pixs[p][t] = x[((size_t)(b * NPIX_ + y * W_ + xx)) * C_ + t];
  }
#pragma unroll
  for (int k = 0; k < 9; ++k) {
    int ny = wy + k / 3 - 1, nx = wx + k % 3 - 1;
    float v = 0.f;
    if (ny >= 0 && ny < HH_ && nx >= 0 && nx < WW_)
      v = wf[((size_t)(b * NW_ + ny * WW_ + nx)) * C_ + t];
    wfus[k][t] = v;
  }
  __syncthreads();
  if (t < 144) {
    int p = t / 9, k = t % 9;
    float s = 0.f;
    for (int c = 0; c < C_; ++c) s += pixs[p][c] * wfus[k][c];
    affs[p][k] = s * 0.0625f;  // scale = C^-0.5
  }
  __syncthreads();
  if (t < PP_) {
    float m = -3.0e38f;
#pragma unroll
    for (int k = 0; k < 9; ++k) m = fmaxf(m, affs[t][k]);
    float e[9], ssum = 0.f;
#pragma unroll
    for (int k = 0; k < 9; ++k) { e[k] = __expf(affs[t][k] - m); ssum += e[k]; }
    float inv = 1.f / ssum;
#pragma unroll
    for (int k = 0; k < 9; ++k) affs[t][k] = e[k] * inv;
  }
  __syncthreads();
  if (t < 144)
    aff[(((size_t)(b * NW_ + n)) * PP_ + t / 9) * 9 + t % 9] = affs[t / 9][t % 9];
#pragma unroll
  for (int k = 0; k < 9; ++k) {
    int ny = wy + k / 3 - 1, nx = wx + k % 3 - 1;
    if (ny < 0 || ny >= HH_ || nx < 0 || nx >= WW_) continue;
    float s = 0.f;
#pragma unroll
    for (int p = 0; p < PP_; ++p) s += pixs[p][t] * affs[p][k];
    atomicAdd(&st[((size_t)(b * NW_ + ny * WW_ + nx)) * C_ + t], s);
    if (t == 0) {
      float mass = 0.f;
#pragma unroll
      for (int p = 0; p < PP_; ++p) mass += affs[p][k];
      atomicAdd(&affsum[b * NW_ + ny * WW_ + nx], mass);
    }
  }
}

// ------------- normalize folded super tokens and convert to f16 -------------
__global__ void k_norm(const float* __restrict__ st, const float* __restrict__ affsum,
                       _Float16* __restrict__ st16) {
  size_t i = (size_t)blockIdx.x * blockDim.x + threadIdx.x;
  st16[i] = (_Float16)(st[i] / (affsum[i / C_] + 1e-12f));
}

__global__ void k_cvt16(const float* __restrict__ in, _Float16* __restrict__ out, int n) {
  int i = blockIdx.x * blockDim.x + threadIdx.x;
  if (i < n) out[i] = (_Float16)in[i];
}

// ----- QKV GEMM: qkv16(b,768,n) = qkv_w16(768,256) x st16(b,n,256) ----------
__global__ void k_gemm_qkv(const _Float16* __restrict__ w16,
                           const _Float16* __restrict__ x16,
                           _Float16* __restrict__ qkv16) {
  int lane = threadIdx.x & 31;
  int N0 = blockIdx.x * 16, O0 = blockIdx.y * 16, b = blockIdx.z;
  int row = O0 + (lane & 15), col = N0 + (lane & 15);
  int akb = (lane < 16) ? 0 : 16, bkb = (lane < 16) ? 0 : 16;
  v8f acc = {0.f, 0.f, 0.f, 0.f, 0.f, 0.f, 0.f, 0.f};
  for (int ks = 0; ks < C_; ks += 32) {
    v16h a, bb;
#pragma unroll
    for (int j = 0; j < 8; ++j) {
      int k0 = 2 * j + akb;
      a[2 * j]     = w16[(size_t)row * C_ + ks + k0];
      a[2 * j + 1] = w16[(size_t)row * C_ + ks + k0 + 1];
      int kb0 = bkb + 2 * j;
      bb[2 * j]     = x16[((size_t)(b * NW_ + col)) * C_ + ks + kb0];
      bb[2 * j + 1] = x16[((size_t)(b * NW_ + col)) * C_ + ks + kb0 + 1];
    }
    acc = WMMA16(a, bb, acc);
  }
#pragma unroll
  for (int r = 0; r < 8; ++r) {
    int o = O0 + r + ((lane < 16) ? 0 : 8);
    qkv16[((size_t)(b * 768 + o)) * NW_ + col] = (_Float16)acc[r];
  }
}

// ---- TDM helper: 2-D tile (rows x 1024 f16, row stride 1024) -> LDS --------
__device__ __forceinline__ void tdm_load_kv(const _Float16* gsrc, void* ldst,
                                            unsigned rows) {
#if __has_builtin(__builtin_amdgcn_tensor_load_to_lds)
  unsigned lds_off = (unsigned)(unsigned long long)ldst;  // flat addr[31:0] = LDS offset
  unsigned long long ga = (unsigned long long)gsrc;
  u32x4 g0;
  g0[0] = 1u;                               // count=1, user mode
  g0[1] = lds_off;                          // lds_addr
  g0[2] = (unsigned)(ga & 0xffffffffu);     // global_addr[31:0]
  g0[3] = (unsigned)((ga >> 32) & 0x01ffffffu) | 0x80000000u;  // addr[56:32] | type=2
  i32x8 g1;
  g1[0] = 1 << 16;                          // data_size=1 (2 bytes), no multicast
  g1[1] = (int)((NW_ & 0xffff) << 16);      // tensor_dim0 = 1024 (lo16 in [31:16])
  g1[2] = (int)((NW_ >> 16) | ((rows & 0xffff) << 16));  // td0 hi, tensor_dim1 lo
  g1[3] = (int)((rows >> 16) | ((NW_ & 0xffff) << 16));  // td1 hi, tile_dim0 = 1024
  g1[4] = (int)(rows & 0xffff);             // tile_dim1 = rows, tile_dim2 = 0
  g1[5] = NW_;                              // tensor_dim0_stride[31:0] = 1024
  g1[6] = 0;                                // stride0 hi, stride1 lo
  g1[7] = 0;
  i32x4 z4 = {0, 0, 0, 0};
  i32x8 z8 = {0, 0, 0, 0, 0, 0, 0, 0};
  // 6-arg form (clang-23 / therock-10.0 headers): groups 2,3 + extra group, cpol
  __builtin_amdgcn_tensor_load_to_lds(g0, g1, z4, z4, z8, 0);
#if __has_builtin(__builtin_amdgcn_s_wait_tensorcnt)
  __builtin_amdgcn_s_wait_tensorcnt(0);
#else
  asm volatile("s_wait_tensorcnt 0x0" ::: "memory");
#endif
#else
  // fallback: cooperative vector copy (K/V blocks are contiguous)
  const unsigned* s = (const unsigned*)gsrc;
  unsigned* d = (unsigned*)ldst;
  int tot = (int)(rows * NW_ / 2);
  for (int i = threadIdx.x; i < tot; i += blockDim.x) d[i] = s[i];
#endif
}

// ----- flash attention over super tokens (softmax over keys), two-pass ------
// K and V for this (b, head) staged once per workgroup into LDS via the
// Tensor Data Mover; WMMA fragments are then fed from LDS.
__global__ void k_attn(const _Float16* __restrict__ qkv16, _Float16* __restrict__ o16) {
  extern __shared__ char smem[];
  _Float16* ks   = (_Float16*)smem;                       // 32 x 1024 f16 (64 KB)
  _Float16* vs   = (_Float16*)(smem + 64 * 1024);         // 32 x 1024 f16 (64 KB)
  _Float16* ldsP = (_Float16*)(smem + 128 * 1024);        // [8][32][17]

  int lane = threadIdx.x & 31, wv = threadIdx.x >> 5;
  int qc = blockIdx.x, head = blockIdx.y, b = blockIdx.z;
  int m0 = qc * 128 + wv * 16;
  int col = lane & 15;
  int akb = (lane < 16) ? 0 : 16, bkb = (lane < 16) ? 0 : 16;
  const _Float16* qp = qkv16 + (size_t)(b * 768 + head * 96) * NW_;
  const _Float16* kp = qp + (size_t)32 * NW_;
  const _Float16* vp = qp + (size_t)64 * NW_;
  const float hs = 0.17677669529663687f;  // 32^-0.5

#if __has_builtin(__builtin_amdgcn_tensor_load_to_lds)
  if (wv == 0) {            // uniform per-wave branch: one wave drives the TDM
    tdm_load_kv(kp, ks, 32);
    tdm_load_kv(vp, vs, 32);
  }
#else
  tdm_load_kv(kp, ks, 32);
  tdm_load_kv(vp, vs, 32);
#endif
  __syncthreads();

  v16h bq;  // Q as B operand: K=d(32) x N=16 queries
#pragma unroll
  for (int j = 0; j < 8; ++j) {
    int d0 = bkb + 2 * j;
    bq[2 * j]     = qp[(size_t)d0 * NW_ + m0 + col];
    bq[2 * j + 1] = qp[(size_t)(d0 + 1) * NW_ + m0 + col];
  }

  // pass 1: max over all 1024 keys per query column
  float mloc = -3.0e38f;
  for (int n0 = 0; n0 < NW_; n0 += 16) {
    v16h ak;
    int krow = n0 + (lane & 15);
#pragma unroll
    for (int j = 0; j < 8; ++j) {
      int d0 = 2 * j + akb;
      ak[2 * j]     = ks[d0 * NW_ + krow];
      ak[2 * j + 1] = ks[(d0 + 1) * NW_ + krow];
    }
    v8f s = {0.f, 0.f, 0.f, 0.f, 0.f, 0.f, 0.f, 0.f};
    s = WMMA16(ak, bq, s);
#pragma unroll
    for (int r = 0; r < 8; ++r) mloc = fmaxf(mloc, s[r] * hs);
  }
  float mcol = fmaxf(mloc, __shfl_xor(mloc, 16, 32));

  // pass 2: P = exp(S - max), sum-exp, O += V x P
  float lpart = 0.f;
  v8f o0 = {0.f, 0.f, 0.f, 0.f, 0.f, 0.f, 0.f, 0.f};
  v8f o1 = {0.f, 0.f, 0.f, 0.f, 0.f, 0.f, 0.f, 0.f};
  for (int n0 = 0; n0 < NW_; n0 += 32) {
#pragma unroll
    for (int sb = 0; sb < 2; ++sb) {
      v16h ak;
      int krow = n0 + sb * 16 + (lane & 15);
#pragma unroll
      for (int j = 0; j < 8; ++j) {
        int d0 = 2 * j + akb;
        ak[2 * j]     = ks[d0 * NW_ + krow];
        ak[2 * j + 1] = ks[(d0 + 1) * NW_ + krow];
      }
      v8f s = {0.f, 0.f, 0.f, 0.f, 0.f, 0.f, 0.f, 0.f};
      s = WMMA16(ak, bq, s);
#pragma unroll
      for (int r = 0; r < 8; ++r) {
        float p = __expf(s[r] * hs - mcol);
        lpart += p;
        int krel = sb * 16 + r + ((lane < 16) ? 0 : 8);
        ldsP[(wv * 32 + krel) * 17 + col] = (_Float16)p;
      }
    }
    asm volatile("s_wait_dscnt 0x0" ::: "memory");
    v16h bp;  // P as B operand: K=32 keys x N=16 queries
#pragma unroll
    for (int j = 0; j < 8; ++j) {
      int k0 = bkb + 2 * j;
      bp[2 * j]     = ldsP[(wv * 32 + k0) * 17 + col];
      bp[2 * j + 1] = ldsP[(wv * 32 + k0 + 1) * 17 + col];
    }
    v16h av0, av1;  // V rows d 0..15 / 16..31 as A operands
    int drow = lane & 15;
#pragma unroll
    for (int j = 0; j < 8; ++j) {
      int k0 = 2 * j + akb;
      av0[2 * j]     = vs[drow * NW_ + n0 + k0];
      av0[2 * j + 1] = vs[drow * NW_ + n0 + k0 + 1];
      av1[2 * j]     = vs[(drow + 16) * NW_ + n0 + k0];
      av1[2 * j + 1] = vs[(drow + 16) * NW_ + n0 + k0 + 1];
    }
    o0 = WMMA16(av0, bp, o0);
    o1 = WMMA16(av1, bp, o1);
    asm volatile("s_wait_dscnt 0x0" ::: "memory");
  }
  float lsum = lpart + __shfl_xor(lpart, 16, 32);
  float inv = 1.f / lsum;
  int m = m0 + col;
#pragma unroll
  for (int r = 0; r < 8; ++r) {
    int d0 = r + ((lane < 16) ? 0 : 8);
    o16[((size_t)(b * NW_ + m)) * C_ + head * HD_ + d0]      = (_Float16)(o0[r] * inv);
    o16[((size_t)(b * NW_ + m)) * C_ + head * HD_ + 16 + d0] = (_Float16)(o1[r] * inv);
  }
}

// ----- proj GEMM + bias: ref(b,n,c) = proj_w(256,256) x attno16(b,n,256) ----
__global__ void k_gemm_proj(const _Float16* __restrict__ w16,
                            const _Float16* __restrict__ x16,
                            const float* __restrict__ bias,
                            float* __restrict__ out) {
  int lane = threadIdx.x & 31;
  int N0 = blockIdx.x * 16, O0 = blockIdx.y * 16, b = blockIdx.z;
  int row = O0 + (lane & 15), col = N0 + (lane & 15);
  int akb = (lane < 16) ? 0 : 16, bkb = (lane < 16) ? 0 : 16;
  v8f acc = {0.f, 0.f, 0.f, 0.f, 0.f, 0.f, 0.f, 0.f};
  for (int ks = 0; ks < C_; ks += 32) {
    v16h a, bb;
#pragma unroll
    for (int j = 0; j < 8; ++j) {
      int k0 = 2 * j + akb;
      a[2 * j]     = w16[(size_t)row * C_ + ks + k0];
      a[2 * j + 1] = w16[(size_t)row * C_ + ks + k0 + 1];
      int kb0 = bkb + 2 * j;
      bb[2 * j]     = x16[((size_t)(b * NW_ + col)) * C_ + ks + kb0];
      bb[2 * j + 1] = x16[((size_t)(b * NW_ + col)) * C_ + ks + kb0 + 1];
    }
    acc = WMMA16(a, bb, acc);
  }
#pragma unroll
  for (int r = 0; r < 8; ++r) {
    int o = O0 + r + ((lane < 16) ? 0 : 8);
    out[((size_t)(b * NW_ + col)) * C_ + o] = acc[r] + bias[o];
  }
}

// ---- scatter refined tokens back: out(b,c,H,W) = unfold3(ref) . aff -------
__global__ void k_scatter(const float* __restrict__ refbuf,
                          const float* __restrict__ aff,
                          float* __restrict__ out) {
  __shared__ float refn[9][C_ + 1];
  __shared__ float affs[PP_][9];
  int n = blockIdx.x, b = blockIdx.y, t = threadIdx.x;
  int wy = n / WW_, wx = n % WW_;
#pragma unroll
  for (int k = 0; k < 9; ++k) {
    int ny = wy + k / 3 - 1, nx = wx + k % 3 - 1;
    float v = 0.f;
    if (ny >= 0 && ny < HH_ && nx >= 0 && nx < WW_)
      v = refbuf[((size_t)(b * NW_ + ny * WW_ + nx)) * C_ + t];
    refn[k][t] = v;
  }
  if (t < 144)
    affs[t / 9][t % 9] = aff[(((size_t)(b * NW_ + n)) * PP_ + t / 9) * 9 + t % 9];
  __syncthreads();
#pragma unroll
  for (int p = 0; p < PP_; ++p) {
    float v = 0.f;
#pragma unroll
    for (int k = 0; k < 9; ++k) v += refn[k][t] * affs[p][k];
    int y = wy * TOK_ + (p >> 2), xx = wx * TOK_ + (p & 3);
    out[((size_t)(b * C_ + t)) * NPIX_ + y * W_ + xx] = v;
  }
}

extern "C" void kernel_launch(void* const* d_in, const int* in_sizes, int n_in,
                              void* d_out, int out_size, void* d_ws, size_t ws_size,
                              hipStream_t stream) {
  const float* x      = (const float*)d_in[0];
  const float* qkv_w  = (const float*)d_in[1];
  const float* proj_w = (const float*)d_in[2];
  const float* proj_b = (const float*)d_in[3];
  float* out = (float*)d_out;

  char* ws = (char*)d_ws;
  size_t off = 0;
  auto alloc = [&](size_t bytes) {
    void* p = ws + off;
    off += (bytes + 255) & ~(size_t)255;
    return p;
  };
  float*    wf      = (float*)alloc((size_t)B_ * NW_ * C_ * 4);          // (b,n,c)
  float*    aff     = (float*)alloc((size_t)B_ * NW_ * PP_ * 9 * 4);     // (b,n,16,9)
  float*    st      = (float*)alloc((size_t)B_ * NW_ * C_ * 4);          // folded wf2
  float*    affsum  = (float*)alloc((size_t)B_ * NW_ * 4);
  _Float16* st16    = (_Float16*)alloc((size_t)B_ * NW_ * C_ * 2);
  _Float16* qkvw16  = (_Float16*)alloc((size_t)768 * C_ * 2);
  _Float16* projw16 = (_Float16*)alloc((size_t)C_ * C_ * 2);
  _Float16* qkv16   = (_Float16*)alloc((size_t)B_ * 768 * NW_ * 2);      // (b,768,n)
  _Float16* attno16 = (_Float16*)alloc((size_t)B_ * NW_ * C_ * 2);       // (b,n,c)
  float*    refbuf  = (float*)alloc((size_t)B_ * NW_ * C_ * 4);          // (b,n,c)

  (void)hipMemsetAsync(st, 0, (size_t)B_ * NW_ * C_ * 4, stream);
  (void)hipMemsetAsync(affsum, 0, (size_t)B_ * NW_ * 4, stream);

  k_pool<<<dim3(NW_, B_), 256, 0, stream>>>(x, wf);
  k_aff<<<dim3(NW_, B_), 256, 0, stream>>>(x, wf, aff, st, affsum);
  k_norm<<<(B_ * NW_ * C_) / 256, 256, 0, stream>>>(st, affsum, st16);
  k_cvt16<<<(768 * C_) / 256, 256, 0, stream>>>(qkv_w, qkvw16, 768 * C_);
  k_cvt16<<<(C_ * C_) / 256, 256, 0, stream>>>(proj_w, projw16, C_ * C_);
  k_gemm_qkv<<<dim3(NW_ / 16, 768 / 16, B_), 32, 0, stream>>>(qkvw16, st16, qkv16);
  // dynamic LDS: K (64K) + V (64K) + P staging (8 waves * 32 * 17 * 2B)
  size_t smem_attn = 128 * 1024 + 8 * 32 * 17 * 2;
  k_attn<<<dim3(NW_ / 128, HEADS_, B_), 256, smem_attn, stream>>>(qkv16, attno16);
  k_gemm_proj<<<dim3(NW_ / 16, C_ / 16, B_), 32, 0, stream>>>(projw16, attno16, proj_b, refbuf);
  k_scatter<<<dim3(NW_, B_), 256, 0, stream>>>(refbuf, aff, out);
}
